// Attention_47682726920277
// MI455X (gfx1250) — matrix operand order repaired and checked by reading, not compiled, not run on hardware
//
#include <hip/hip_runtime.h>
#include <hip/hip_bf16.h>

typedef _Float16 half_t;
typedef __attribute__((ext_vector_type(16))) _Float16 v16h;
typedef __attribute__((ext_vector_type(8)))  _Float16 v8h;
typedef __attribute__((ext_vector_type(8)))  float    v8f;

#define DD   1024
#define TT   2048
#define HH   16
#define HD   64
#define BB   2
#define SROW 2052   // padded f32 row stride in LDS (avoids bank conflicts)

__device__ __forceinline__ v8f wmma_f16(v16h a, v16h b, v8f c) {
    return __builtin_amdgcn_wmma_f32_16x16x32_f16(
        /*neg_a=*/false, a, /*neg_b=*/false, b,
        /*c_mod=*/(short)0, c, /*reuse_a=*/false, /*reuse_b=*/false);
}

// A-matrix fragment (16x32, f16). ISA layout:
// lanes 0-15 : row M=lane,    elems 0..7 = K0..7,  elems 8..15 = K16..23
// lanes 16-31: row M=lane-16, elems 0..7 = K8..15, elems 8..15 = K24..31
__device__ __forceinline__ v16h load_fragA(const half_t* __restrict__ base, int ld) {
    int lane = threadIdx.x & 31;
    int r = lane & 15, hf = lane >> 4;
    const half_t* p = base + (size_t)r * ld + hf * 8;
    v8h lo = *(const v8h*)(p);
    v8h hi = *(const v8h*)(p + 16);
    v16h out;
    #pragma unroll
    for (int i = 0; i < 8; ++i) { out[i] = lo[i]; out[i + 8] = hi[i]; }
    return out;
}

// Same A layout but source row lives in LDS (P matrix after softmax).
__device__ __forceinline__ v16h load_fragA_lds(const half_t* p_base) {
    v8h lo = *(const v8h*)(p_base);
    v8h hi = *(const v8h*)(p_base + 16);
    v16h out;
    #pragma unroll
    for (int i = 0; i < 8; ++i) { out[i] = lo[i]; out[i + 8] = hi[i]; }
    return out;
}

// B-matrix fragment (32x16, f16). Column n = row n of the B-source matrix.
// lanes 0-15 : col N=lane,    elems 0..7 = K0..7,  elems 8..15 = K16..23
// lanes 16-31: col N=lane-16, elems 0..7 = K8..15, elems 8..15 = K24..31
__device__ __forceinline__ v16h load_fragB(const half_t* __restrict__ base, int ld) {
    int lane = threadIdx.x & 31;
    int n = lane & 15, hf = lane >> 4;
    const half_t* p = base + (size_t)n * ld + hf * 8;
    v8h lo = *(const v8h*)(p);
    v8h hi = *(const v8h*)(p + 16);
    v16h out;
    #pragma unroll
    for (int i = 0; i < 8; ++i) { out[i] = lo[i]; out[i + 8] = hi[i]; }
    return out;
}

__global__ void cvt_f32_f16(const float* __restrict__ in, half_t* __restrict__ out, int n) {
    int i = blockIdx.x * blockDim.x + threadIdx.x;
    if (i < n) out[i] = (half_t)in[i];
}

// One wave computes a 16x64 tile of y = A[16,1024] @ B^T where B rows are K-contiguous.
// mode 0: store f32 to outF[m*1024+n]
// mode 1: RoPE epilogue -> outH as [B,H,T,64] f16   (Q / K)
// mode 2: transpose epilogue -> outH as [B,H,64,T] f16  (V)
__global__ __launch_bounds__(32) void gemm_qkvo(
    const half_t* __restrict__ A, const half_t* __restrict__ Bm,
    float* __restrict__ outF, half_t* __restrict__ outH,
    const float* __restrict__ cosT, const float* __restrict__ sinT, int mode)
{
    int lane = threadIdx.x & 31;
    int r16 = lane & 15, hf = lane >> 4;
    int n0 = blockIdx.x * 64;
    int m0 = blockIdx.y * 16;

    v8f acc[4] = {v8f{}, v8f{}, v8f{}, v8f{}};
    const half_t* arow = A + (size_t)m0 * DD;
    for (int k0 = 0; k0 < DD; k0 += 32) {
        v16h af = load_fragA(arow + k0, DD);
        #pragma unroll
        for (int t = 0; t < 4; ++t) {
            v16h bf = load_fragB(Bm + (size_t)(n0 + 16 * t) * DD + k0, DD);
            acc[t] = wmma_f16(af, bf, acc[t]);
        }
    }

    if (mode == 0) {
        #pragma unroll
        for (int t = 0; t < 4; ++t)
            #pragma unroll
            for (int i = 0; i < 8; ++i) {
                int m = m0 + (hf << 3) + i;
                int n = n0 + 16 * t + r16;
                outF[(size_t)m * DD + n] = acc[t][i];
            }
    } else if (mode == 1) {
        #pragma unroll
        for (int t = 0; t < 4; ++t)
            #pragma unroll
            for (int i = 0; i < 8; ++i) {
                int m = m0 + (hf << 3) + i;
                int n = n0 + 16 * t + r16;
                int b = m >> 11, tp = m & (TT - 1);
                int h = n >> 6,  d = n & (HD - 1);
                float c = cosT[tp * HD + d];
                float s = sinT[tp * HD + d];
                float rot = (t < 2) ? -acc[t + 2][i] : acc[t - 2][i];
                float val = acc[t][i] * c + rot * s;
                outH[(((size_t)(b * HH + h) * TT) + tp) * HD + d] = (half_t)val;
            }
    } else {
        #pragma unroll
        for (int t = 0; t < 4; ++t)
            #pragma unroll
            for (int i = 0; i < 8; ++i) {
                int m = m0 + (hf << 3) + i;
                int n = n0 + 16 * t + r16;
                int b = m >> 11, tp = m & (TT - 1);
                int h = n >> 6,  d = n & (HD - 1);
                outH[(((size_t)(b * HH + h) * HD) + d) * TT + tp] = (half_t)acc[t][i];
            }
    }
}

// One wave per (b,h, 16-query block). Scores staged in LDS (16 x 2048 f32, padded),
// causal softmax lane-per-row, then P@V with WMMA reading P from LDS.
__global__ __launch_bounds__(32) void attn_core(
    const half_t* __restrict__ q, const half_t* __restrict__ k,
    const half_t* __restrict__ vT, half_t* __restrict__ o)
{
    extern __shared__ float S[];  // [16][SROW]
    int lane = threadIdx.x & 31;
    int r16 = lane & 15, hf = lane >> 4;
    int m0 = blockIdx.x * 16;
    int bh = blockIdx.y;
    int b = bh >> 4, h = bh & 15;

    const half_t* qb = q  + ((size_t)bh * TT + m0) * HD;
    const half_t* kb = k  + (size_t)bh * TT * HD;
    const half_t* vb = vT + (size_t)bh * HD * TT;

    v16h aq0 = load_fragA(qb, HD);       // d = 0..31
    v16h aq1 = load_fragA(qb + 32, HD);  // d = 32..63
    const int jend = m0 + 16;            // keys needed: [0, jend)
    const float scale = 0.125f;          // 1/sqrt(64)

    // Pass 1: scores -> LDS with causal mask.
    for (int j0 = 0; j0 < TT; j0 += 32) {
        v8f s0{}, s1{};
        bool active = (j0 < jend);       // uniform across the wave
        if (active) {
            v16h b0a = load_fragB(kb + (size_t)j0 * HD, HD);
            v16h b0b = load_fragB(kb + (size_t)j0 * HD + 32, HD);
            s0 = wmma_f16(aq0, b0a, s0);
            s0 = wmma_f16(aq1, b0b, s0);
            v16h b1a = load_fragB(kb + (size_t)(j0 + 16) * HD, HD);
            v16h b1b = load_fragB(kb + (size_t)(j0 + 16) * HD + 32, HD);
            s1 = wmma_f16(aq0, b1a, s1);
            s1 = wmma_f16(aq1, b1b, s1);
        }
        #pragma unroll
        for (int i = 0; i < 8; ++i) {
            int row = (hf << 3) + i;
            int gq = m0 + row;
            int j = j0 + r16;
            float v0 = active ? s0[i] * scale : -INFINITY;
            if (j > gq) v0 = -INFINITY;
            S[row * SROW + j] = v0;
            float v1 = active ? s1[i] * scale : -INFINITY;
            if (j + 16 > gq) v1 = -INFINITY;
            S[row * SROW + j + 16] = v1;
        }
    }
    __syncthreads();

    // Pass 2: softmax, lane r owns row r; compact to f16 in place.
    if (lane < 16) {
        float* row = S + lane * SROW;
        float mx = -INFINITY;
        for (int e = 0; e < TT; ++e) mx = fmaxf(mx, row[e]);
        float sum = 0.f;
        for (int e = 0; e < TT; ++e) { float ex = __expf(row[e] - mx); row[e] = ex; sum += ex; }
        float inv = 1.0f / sum;
        half_t* rh = (half_t*)row;
        for (int e = 0; e < TT; ++e) rh[e] = (half_t)(row[e] * inv);  // safe in-place
    }
    __syncthreads();

    // Pass 3: out[16,64] = P @ V  (A from LDS, B from vT rows = contiguous j).
    v8f acc[4] = {v8f{}, v8f{}, v8f{}, v8f{}};
    for (int j0 = 0; j0 < jend; j0 += 32) {
        const half_t* prow = (const half_t*)(S + r16 * SROW);
        v16h ap = load_fragA_lds(prow + j0 + hf * 8);
        #pragma unroll
        for (int t = 0; t < 4; ++t) {
            v16h bv = load_fragB(vb + (size_t)(16 * t) * TT + j0, TT);
            acc[t] = wmma_f16(ap, bv, acc[t]);
        }
    }

    #pragma unroll
    for (int t = 0; t < 4; ++t)
        #pragma unroll
        for (int i = 0; i < 8; ++i) {
            int row = (hf << 3) + i;
            int tp = m0 + row;
            int d = 16 * t + r16;
            o[(((size_t)b * TT + tp) * DD) + h * HD + d] = (half_t)acc[t][i];
        }
}

extern "C" void kernel_launch(void* const* d_in, const int* in_sizes, int n_in,
                              void* d_out, int out_size, void* d_ws, size_t ws_size,
                              hipStream_t stream) {
    const float* x    = (const float*)d_in[0];
    const float* cosT = (const float*)d_in[1];
    const float* sinT = (const float*)d_in[2];
    // d_in[3] = mask (implied by causal indexing; unused)
    const float* wq   = (const float*)d_in[4];
    const float* wk   = (const float*)d_in[5];
    const float* wv   = (const float*)d_in[6];
    const float* wo   = (const float*)d_in[7];
    float* out = (float*)d_out;

    char* ws = (char*)d_ws;
    const size_t XN = (size_t)BB * TT * DD;   // 4M elements
    const size_t WN = (size_t)DD * DD;        // 1M elements
    half_t* xh  = (half_t*)(ws);                                   // 8 MB (also reused as attn buf)
    half_t* wqh = (half_t*)(ws + XN * 2);                          // 2 MB
    half_t* wkh = wqh + WN;
    half_t* wvh = wkh + WN;
    half_t* woh = wvh + WN;
    half_t* qh  = (half_t*)(ws + XN * 2 + 4 * WN * 2);             // 8 MB [B,H,T,64]
    half_t* kh  = qh + XN;                                         // 8 MB
    half_t* vTh = kh + XN;                                         // 8 MB [B,H,64,T]
    half_t* attn = xh;  // x is consumed by QKV GEMMs; reuse as attention output [B,T,D]

    // f32 -> f16 conversions
    cvt_f32_f16<<<(int)((XN + 255) / 256), 256, 0, stream>>>(x,  xh,  (int)XN);
    cvt_f32_f16<<<(int)((WN + 255) / 256), 256, 0, stream>>>(wq, wqh, (int)WN);
    cvt_f32_f16<<<(int)((WN + 255) / 256), 256, 0, stream>>>(wk, wkh, (int)WN);
    cvt_f32_f16<<<(int)((WN + 255) / 256), 256, 0, stream>>>(wv, wvh, (int)WN);
    cvt_f32_f16<<<(int)((WN + 255) / 256), 256, 0, stream>>>(wo, woh, (int)WN);

    dim3 ggrid(DD / 64, (BB * TT) / 16);  // (16, 256)
    gemm_qkvo<<<ggrid, 32, 0, stream>>>(xh, wqh, nullptr, qh,  cosT, sinT, 1); // Q + RoPE
    gemm_qkvo<<<ggrid, 32, 0, stream>>>(xh, wkh, nullptr, kh,  cosT, sinT, 1); // K + RoPE
    gemm_qkvo<<<ggrid, 32, 0, stream>>>(xh, wvh, nullptr, vTh, cosT, sinT, 2); // V transposed

    dim3 agrid(TT / 16, BB * HH);         // (128, 32)
    size_t lds = (size_t)16 * SROW * sizeof(float);  // ~131 KB dynamic LDS
    attn_core<<<agrid, 32, lds, stream>>>(qh, kh, vTh, attn);

    gemm_qkvo<<<ggrid, 32, 0, stream>>>(attn, woh, out, nullptr, cosT, sinT, 0); // output proj -> f32
}
